// EALSTM_68925635166882
// MI455X (gfx1250) — compile-verified, matrix operand order checked by reading, not run on hardware
//
#include <hip/hip_runtime.h>
#include <hip/hip_bf16.h>

typedef __bf16 bf16;
typedef __attribute__((ext_vector_type(16))) __bf16 v16bf;
typedef __attribute__((ext_vector_type(8)))  __bf16 v8bf;
typedef __attribute__((ext_vector_type(8)))  float  v8f;

#define B_   256
#define T_   365
#define I_   32
#define H_   256
#define S_   27
#define L_   7
#define F_   8

#define MTILE  32           // batch rows per workgroup (2 WMMA M-tiles)
#define STRIDE 296          // padded LDS row stride in bf16 (592B, 16B aligned, bank-conflict free)
#define KTOT   288          // K = H (256) + I (32)

// bf16 workspace layout (element offsets)
#define W0_OFF 0
#define W1_OFF (768*288)
#define WP_OFF (2*768*288)
#define WS_ELEMS (2*768*288 + 32*256)

__device__ __forceinline__ float sigmoidf_(float x) { return 1.0f / (1.0f + __expf(-x)); }
__device__ __forceinline__ float tanhf_(float x) {
  float e = __expf(-2.0f * x);
  return (1.0f - e) / (1.0f + e);
}

__device__ __forceinline__ v8f wmma_bf16(v16bf a, v16bf b, v8f c) {
  return __builtin_amdgcn_wmma_f32_16x16x32_bf16(false, a, false, b, (short)0, c, false, false);
}

// 16x32 bf16 fragment from row-major [16][row_stride] tile, K-tile at `kbase`.
// CDNA5 16-bit A/B layout: lanes 0-15: row=lane, K=kbase+{0..7,16..23};
// lanes 16-31: row=lane-16, K=kbase+{8..15,24..31}. Two 16B loads per lane.
// Hoisting out of the timestep loop is blocked by a memory-clobber asm at the
// top of each iteration (not by volatile, which would serialize at SYS scope).
__device__ __forceinline__ v16bf load_frag16(const bf16* base, int row_stride, int kbase) {
  int lane = threadIdx.x & 31;
  int row = lane & 15;
  int ko = kbase + ((lane >> 4) << 3);
  const bf16* p = base + row * row_stride + ko;
  v8bf lo = *(const v8bf*)(p);
  v8bf hi = *(const v8bf*)(p + 16);
  v16bf r;
#pragma unroll
  for (int i = 0; i < 8; ++i) { r[i] = lo[i]; r[i + 8] = hi[i]; }
  return r;
}

// ---------------------------------------------------------------------------
// Pack fp32 weights -> bf16 row-major workspace.
// w0/w1: [768][288] = [Whg | Wdg] per layer.  wp: [32][256] = Wp0.
// ---------------------------------------------------------------------------
__global__ void ealstm_pack_kernel(const float* __restrict__ Whg0, const float* __restrict__ Wdg0,
                                   const float* __restrict__ Whg1, const float* __restrict__ Wdg1,
                                   const float* __restrict__ Wp0, bf16* __restrict__ ws) {
  int idx = blockIdx.x * blockDim.x + threadIdx.x;
  if (idx >= WS_ELEMS) return;
  float v;
  if (idx < W1_OFF) {
    int n = idx / 288, k = idx % 288;
    v = (k < 256) ? Whg0[n * 256 + k] : Wdg0[n * 32 + (k - 256)];
  } else if (idx < WP_OFF) {
    int j = idx - W1_OFF;
    int n = j / 288, k = j % 288;
    v = (k < 256) ? Whg1[n * 256 + k] : Wdg1[n * 32 + (k - 256)];
  } else {
    v = Wp0[idx - WP_OFF];
  }
  ws[idx] = (bf16)v;
}

// ---------------------------------------------------------------------------
// Persistent fused EA-LSTM: 8 workgroups x 32 batch rows, 256 threads (8 waves).
// Wave w owns hidden columns [32w, 32w+32): 6 gate n-tiles (f/o/g x 2), each
// weight fragment feeds two M-tiles (2x arithmetic intensity vs L2).
// ---------------------------------------------------------------------------
__global__ void __launch_bounds__(256)
ealstm_main_kernel(const float* __restrict__ x, const float* __restrict__ stat,
                   const float* __restrict__ fut,
                   const float* __restrict__ Wsh0, const float* __restrict__ bsh0,
                   const float* __restrict__ bdg0,
                   const float* __restrict__ Wsh1, const float* __restrict__ bsh1,
                   const float* __restrict__ bdg1,
                   const float* __restrict__ bp0,
                   const float* __restrict__ Wh1, const float* __restrict__ bh1,
                   const float* __restrict__ Wh2, const float* __restrict__ bh2,
                   const float* __restrict__ Wf1, const float* __restrict__ bf1,
                   const float* __restrict__ Wf2, const float* __restrict__ bf2,
                   const bf16* __restrict__ wsw, float* __restrict__ out) {
  __shared__ bf16 hx0[32 * STRIDE];   // [h0 (256) | x_t (32)] bf16, 32 rows
  __shared__ bf16 hx1[32 * STRIDE];   // [h1 (256) | inp1 (32)] bf16, 32 rows
  __shared__ float tmp[16 * 256];     // head hidden scratch (reused)

  const int tid = threadIdx.x;
  const int wv = tid >> 5;
  const int lane = tid & 31;
  const int b0 = blockIdx.x * MTILE;

  const bf16* w0 = wsw + W0_OFF;
  const bf16* w1 = wsw + W1_OFF;
  const bf16* wp = wsw + WP_OFF;

  const int col0 = wv * 32;           // this wave's hidden-column base
  const int nlocal = lane & 15;       // D-frag column within tile
  const int mrow = (lane >> 4) * 8;   // D-frag row base within M-tile

  // Biases per (layer, tile): column-only, one scalar per lane.
  float bF[2][2], bO[2][2], bG[2][2];
#pragma unroll
  for (int j = 0; j < 2; ++j) {
    int n = col0 + 16 * j + nlocal;
    bF[0][j] = bdg0[n]; bO[0][j] = bdg0[256 + n]; bG[0][j] = bdg0[512 + n];
    bF[1][j] = bdg1[n]; bO[1][j] = bdg1[256 + n]; bG[1][j] = bdg1[512 + n];
  }

  // Static input gates + cell state, in D-fragment register layout: [tile j][mtile][row].
  float ig0[2][2][8], ig1[2][2][8], cst0[2][2][8], cst1[2][2][8];
#pragma unroll
  for (int j = 0; j < 2; ++j) {
    int n = col0 + 16 * j + nlocal;
#pragma unroll
    for (int mt = 0; mt < 2; ++mt)
#pragma unroll
      for (int r = 0; r < 8; ++r) {
        int b = b0 + mt * 16 + mrow + r;
        float s0 = bsh0[n], s1 = bsh1[n];
        for (int s = 0; s < S_; ++s) {
          float sv = stat[b * S_ + s];
          s0 += sv * Wsh0[n * S_ + s];
          s1 += sv * Wsh1[n * S_ + s];
        }
        ig0[j][mt][r] = sigmoidf_(s0);
        ig1[j][mt][r] = sigmoidf_(s1);
        cst0[j][mt][r] = 0.0f;
        cst1[j][mt][r] = 0.0f;
      }
  }

  // Zero h parts; load x_0 into hx0 x-part.
  {
#pragma unroll 1
    for (int e = 0; e < 32; ++e) {
      int idx = tid + 256 * e;          // 8192 = 32 rows x 256 cols
      int m = idx >> 8, n = idx & 255;
      hx0[m * STRIDE + n] = (bf16)0.0f;
      hx1[m * STRIDE + n] = (bf16)0.0f;
    }
    int i = tid & 31, mq = tid >> 5;
#pragma unroll
    for (int e = 0; e < 4; ++e) {
      int m = mq + 8 * e;
      hx0[m * STRIDE + 256 + i] = (bf16)x[(size_t)(b0 + m) * T_ * I_ + i];
    }
  }
  __syncthreads();

#pragma unroll 1
  for (int t = 0; t < T_; ++t) {
    // Block LICM of the (loop-invariant) weight loads: treated as a potential
    // store to any location, so weights re-stream from L2 every timestep with
    // plain default-scope global_load_b128 (clause-able, pipelined).
    asm volatile("" ::: "memory");

    // -------- Phase A: layer-0 gates = [h0|x_t] @ [Whg0|Wdg0]^T --------
    float hnew0[2][2][8];
    {
      v8f acc[6][2] = {};
#pragma unroll
      for (int kt = 0; kt < 9; ++kt) {
        v16bf a0 = load_frag16(hx0, STRIDE, kt * 32);
        v16bf a1 = load_frag16(hx0 + 16 * STRIDE, STRIDE, kt * 32);
#pragma unroll
        for (int q = 0; q < 6; ++q) {
          int part = q >> 1, j = q & 1;
          int ntile = part * 16 + (col0 >> 4) + j;
          v16bf b = load_frag16(w0 + ntile * 16 * KTOT, KTOT, kt * 32);
          acc[q][0] = wmma_bf16(a0, b, acc[q][0]);
          acc[q][1] = wmma_bf16(a1, b, acc[q][1]);
        }
      }
#pragma unroll
      for (int j = 0; j < 2; ++j)
#pragma unroll
        for (int mt = 0; mt < 2; ++mt)
#pragma unroll
          for (int r = 0; r < 8; ++r) {
            float f = sigmoidf_(acc[0 + j][mt][r] + bF[0][j]);
            float o = sigmoidf_(acc[2 + j][mt][r] + bO[0][j]);
            float g = tanhf_(acc[4 + j][mt][r] + bG[0][j]);
            float c = f * cst0[j][mt][r] + ig0[j][mt][r] * g;
            cst0[j][mt][r] = c;
            hnew0[j][mt][r] = o * tanhf_(c);
          }
    }
    __syncthreads();
    // write h0_new + x_{t+1}
#pragma unroll
    for (int j = 0; j < 2; ++j)
#pragma unroll
      for (int mt = 0; mt < 2; ++mt)
#pragma unroll
        for (int r = 0; r < 8; ++r)
          hx0[(mt * 16 + mrow + r) * STRIDE + col0 + 16 * j + nlocal] = (bf16)hnew0[j][mt][r];
    if (t + 1 < T_) {
      int i = tid & 31, mq = tid >> 5;
#pragma unroll
      for (int e = 0; e < 4; ++e) {
        int m = mq + 8 * e;
        hx0[m * STRIDE + 256 + i] =
            (bf16)x[(size_t)(b0 + m) * T_ * I_ + (size_t)(t + 1) * I_ + i];
      }
    }
    __syncthreads();

    // -------- Phase B: inp1 = h0_new @ Wp0^T + bp0 (waves 0..3) --------
    if (wv < 4) {
      int mt = wv >> 1, nt = wv & 1;
      v8f acc = {};
#pragma unroll
      for (int kt = 0; kt < 8; ++kt) {
        v16bf a = load_frag16(hx0 + mt * 16 * STRIDE, STRIDE, kt * 32);
        v16bf b = load_frag16(wp + nt * 16 * 256, 256, kt * 32);
        acc = wmma_bf16(a, b, acc);
      }
      float bp = bp0[nt * 16 + nlocal];
#pragma unroll
      for (int r = 0; r < 8; ++r)
        hx1[(mt * 16 + mrow + r) * STRIDE + 256 + nt * 16 + nlocal] = (bf16)(acc[r] + bp);
    }
    __syncthreads();

    // -------- Phase C: layer-1 gates = [h1|inp1] @ [Whg1|Wdg1]^T --------
    float hnew1[2][2][8];
    {
      v8f acc[6][2] = {};
#pragma unroll
      for (int kt = 0; kt < 9; ++kt) {
        v16bf a0 = load_frag16(hx1, STRIDE, kt * 32);
        v16bf a1 = load_frag16(hx1 + 16 * STRIDE, STRIDE, kt * 32);
#pragma unroll
        for (int q = 0; q < 6; ++q) {
          int part = q >> 1, j = q & 1;
          int ntile = part * 16 + (col0 >> 4) + j;
          v16bf b = load_frag16(w1 + ntile * 16 * KTOT, KTOT, kt * 32);
          acc[q][0] = wmma_bf16(a0, b, acc[q][0]);
          acc[q][1] = wmma_bf16(a1, b, acc[q][1]);
        }
      }
#pragma unroll
      for (int j = 0; j < 2; ++j)
#pragma unroll
        for (int mt = 0; mt < 2; ++mt)
#pragma unroll
          for (int r = 0; r < 8; ++r) {
            float f = sigmoidf_(acc[0 + j][mt][r] + bF[1][j]);
            float o = sigmoidf_(acc[2 + j][mt][r] + bO[1][j]);
            float g = tanhf_(acc[4 + j][mt][r] + bG[1][j]);
            float c = f * cst1[j][mt][r] + ig1[j][mt][r] * g;
            cst1[j][mt][r] = c;
            hnew1[j][mt][r] = o * tanhf_(c);
          }
    }
    __syncthreads();
#pragma unroll
    for (int j = 0; j < 2; ++j)
#pragma unroll
      for (int mt = 0; mt < 2; ++mt)
#pragma unroll
        for (int r = 0; r < 8; ++r)
          hx1[(mt * 16 + mrow + r) * STRIDE + col0 + 16 * j + nlocal] = (bf16)hnew1[j][mt][r];
    __syncthreads();
  }

  // -------- Heads, processed in two 16-row halves reusing one tmp buffer ------
#pragma unroll 1
  for (int half = 0; half < 2; ++half) {
    int mb = half * 16;
    // stage 1: projection head hidden = relu(h1 @ Wh1^T + bh1)
    {
      int n = tid;
      for (int m = 0; m < 16; ++m) {
        float s = bh1[n];
        for (int k = 0; k < 256; ++k)
          s += (float)hx1[(mb + m) * STRIDE + k] * Wh1[n * 256 + k];
        tmp[m * 256 + n] = fmaxf(s, 0.0f);
      }
    }
    __syncthreads();
    float s1 = 0.0f;
    int om = tid / L_, ol = tid % L_;
    if (tid < 16 * L_) {
      s1 = bh2[ol];
      for (int k = 0; k < 256; ++k) s1 += tmp[om * 256 + k] * Wh2[ol * 256 + k];
    }
    __syncthreads();
    // stage 2: future head hidden = relu(future @ Wf1^T + bf1)
    {
      int n = tid;
      for (int m = 0; m < 16; ++m) {
        float s = bf1[n];
        const float* fr = fut + (size_t)(b0 + mb + m) * (F_ * L_);
        for (int k = 0; k < F_ * L_; ++k)
          s += fr[k] * Wf1[n * (F_ * L_) + k];
        tmp[m * 256 + n] = fmaxf(s, 0.0f);
      }
    }
    __syncthreads();
    if (tid < 16 * L_) {
      float s2 = bf2[ol];
      for (int k = 0; k < 256; ++k) s2 += tmp[om * 256 + k] * Wf2[ol * 256 + k];
      out[(size_t)(b0 + mb + om) * L_ + ol] = s1 + s2;
    }
    __syncthreads();
  }
}

extern "C" void kernel_launch(void* const* d_in, const int* in_sizes, int n_in,
                              void* d_out, int out_size, void* d_ws, size_t ws_size,
                              hipStream_t stream) {
  (void)in_sizes; (void)n_in; (void)out_size; (void)ws_size;
  const float* x      = (const float*)d_in[0];
  const float* stat   = (const float*)d_in[1];
  const float* fut    = (const float*)d_in[2];
  const float* Wsh0   = (const float*)d_in[3];
  const float* bsh0   = (const float*)d_in[4];
  const float* Wdg0   = (const float*)d_in[5];
  const float* bdg0   = (const float*)d_in[6];
  const float* Whg0   = (const float*)d_in[7];
  const float* Wsh1   = (const float*)d_in[8];
  const float* bsh1   = (const float*)d_in[9];
  const float* Wdg1   = (const float*)d_in[10];
  const float* bdg1   = (const float*)d_in[11];
  const float* Whg1   = (const float*)d_in[12];
  const float* Wp0    = (const float*)d_in[13];
  const float* bp0    = (const float*)d_in[14];
  const float* Wh1    = (const float*)d_in[15];
  const float* bh1    = (const float*)d_in[16];
  const float* Wh2    = (const float*)d_in[17];
  const float* bh2    = (const float*)d_in[18];
  const float* Wf1    = (const float*)d_in[19];
  const float* bf1    = (const float*)d_in[20];
  const float* Wf2    = (const float*)d_in[21];
  const float* bf2    = (const float*)d_in[22];

  bf16* ws = (bf16*)d_ws;
  float* out = (float*)d_out;

  int pack_blocks = (WS_ELEMS + 255) / 256;
  ealstm_pack_kernel<<<pack_blocks, 256, 0, stream>>>(Whg0, Wdg0, Whg1, Wdg1, Wp0, ws);

  ealstm_main_kernel<<<B_ / MTILE, 256, 0, stream>>>(
      x, stat, fut, Wsh0, bsh0, bdg0, Wsh1, bsh1, bdg1, bp0,
      Wh1, bh1, Wh2, bh2, Wf1, bf1, Wf2, bf2, ws, out);
}
